// MixtralSparseMoeBlock_8976481649134
// MI455X (gfx1250) — matrix operand order repaired and checked by reading, not compiled, not run on hardware
//
#include <hip/hip_runtime.h>
#include <hip/hip_bf16.h>

// ---------------- problem constants (fixed by the reference) ----------------
#define E_        32
#define C_        512
#define T_        2048
#define K_        2
#define H_        1024
#define F_        2816
#define NA_       4096      // T_*K_ assignments
#define ROWS_CAP  5120      // 4096 + 32 experts * 31 pad, rounded up

typedef __attribute__((ext_vector_type(16))) __bf16 v16bf;
typedef __attribute__((ext_vector_type(8)))  float  v8f;

// f32 -> bf16 round-to-nearest-even (epilogue only; cold path)
__device__ __forceinline__ __bf16 f2bf(float f) {
  union { float f; unsigned u; } x; x.f = f;
  unsigned r = x.u + 0x7FFFu + ((x.u >> 16) & 1u);
  unsigned short h = (unsigned short)(r >> 16);
  __bf16 b;
  __builtin_memcpy(&b, &h, 2);
  return b;
}

// Pack two f32 into two bf16 (truncation) with a single v_perm_b32:
// result = { hi16(hi), hi16(lo) }.  Byte sel: 0,1 <- lo[2],lo[3]; 2,3 <- hi[2],hi[3]
__device__ __forceinline__ unsigned pk2bf(float lo, float hi) {
  union { float f; unsigned u; } a, b;
  a.f = lo; b.f = hi;
  return __builtin_amdgcn_perm(b.u, a.u, 0x07060302u);
}

union V16A { v16bf v; int4 q[2]; };
union V16D { v16bf v; unsigned d[8]; };

// A fragment (16x32 bf16), contiguous k order (same as B):
//   lane: half=lane>>4, m=lane&15; j=0..15 -> K = kb + half*16 + j
__device__ __forceinline__ v16bf load_a_frag(const __bf16* __restrict__ A, int row0,
                                             int ldk, int kb, int lane) {
  const int half = lane >> 4, m = lane & 15;
  const __bf16* p = A + (size_t)(row0 + m) * ldk + kb + half * 16;
  V16A u;
  u.q[0] = *(const int4*)(p);
  u.q[1] = *(const int4*)(p + 8);
  return u.v;
}

// B fragment (32x16 bf16) for C = A @ W^T with row-major W[N][K]:
//   lane n = column, j=0..15 -> K = kb + half*16 + j  (16 contiguous f32 per lane)
//   8x v_perm_b32 does the whole f32->bf16 conversion for the fragment.
__device__ __forceinline__ v16bf load_b_frag(const float* __restrict__ W, int col0,
                                             int ldk, int kb, int lane) {
  const int half = lane >> 4, n = lane & 15;
  const float4* p = (const float4*)(W + (size_t)(col0 + n) * ldk + kb + half * 16);
  float4 f0 = p[0], f1 = p[1], f2 = p[2], f3 = p[3];
  V16D u;
  u.d[0] = pk2bf(f0.x, f0.y);
  u.d[1] = pk2bf(f0.z, f0.w);
  u.d[2] = pk2bf(f1.x, f1.y);
  u.d[3] = pk2bf(f1.z, f1.w);
  u.d[4] = pk2bf(f2.x, f2.y);
  u.d[5] = pk2bf(f2.z, f2.w);
  u.d[6] = pk2bf(f3.x, f3.y);
  u.d[7] = pk2bf(f3.z, f3.w);
  return u.v;
}

__device__ __forceinline__ v8f wmma_bf16(v16bf a, v16bf b, v8f c) {
  return __builtin_amdgcn_wmma_f32_16x16x32_bf16(false, a, false, b, (short)0, c,
                                                 false, false);
}

// ---------------------------------------------------------------------------
// 1) Router: logits = x @ gate_w^T ; softmax(fp32); top-k weights normalized.
// ---------------------------------------------------------------------------
__global__ __launch_bounds__(128) void router_kernel(
    const float* __restrict__ x, const float* __restrict__ gate_w,
    const int* __restrict__ sel, float* __restrict__ logits_out,
    float* __restrict__ selw) {
  __shared__ float xs[H_];
  __shared__ float part[128];
  __shared__ float lg[E_];
  __shared__ float red[2];
  const int t = blockIdx.x, tid = threadIdx.x;
  const float* xr = x + (size_t)t * H_;
  for (int i = tid; i < H_; i += 128) xs[i] = xr[i];
  __syncthreads();
  const int e = tid & 31, seg = tid >> 5;           // 4 segments of 256
  const float* g = gate_w + (size_t)e * H_ + seg * 256;
  float s = 0.f;
#pragma unroll 8
  for (int j = 0; j < 256; ++j) s += xs[seg * 256 + j] * g[j];
  part[tid] = s;
  __syncthreads();
  if (tid < E_) {
    float l = part[tid] + part[tid + 32] + part[tid + 64] + part[tid + 96];
    lg[tid] = l;
    logits_out[t * E_ + tid] = l;
  }
  __syncthreads();
  if (tid == 0) {
    float mx = lg[0];
    for (int i = 1; i < E_; ++i) mx = fmaxf(mx, lg[i]);
    float d = 0.f;
    for (int i = 0; i < E_; ++i) d += __expf(lg[i] - mx);
    red[0] = mx; red[1] = d;
  }
  __syncthreads();
  if (tid < K_) {
    int e0 = sel[t * K_ + 0], e1 = sel[t * K_ + 1];
    float p0 = __expf(lg[e0] - red[0]) / red[1];
    float p1 = __expf(lg[e1] - red[0]) / red[1];
    float pk = (tid == 0) ? p0 : p1;
    selw[t * K_ + tid] = pk / (p0 + p1);
  }
}

// ---------------------------------------------------------------------------
// 2) Dispatch: stable-sort-equivalent ranking. Wave w owns expert w.
//    base[e] padded to 32 rows so 32-row M-tiles never cross experts.
// ---------------------------------------------------------------------------
__global__ __launch_bounds__(1024) void dispatch_kernel(
    const int* __restrict__ sel, int* __restrict__ cnt, int* __restrict__ basep,
    int* __restrict__ row_tok, int* __restrict__ row_assign) {
  __shared__ int eids[NA_];
  __shared__ int s_cnt[E_];
  __shared__ int s_base[E_];
  const int tid = threadIdx.x;
  for (int i = tid; i < NA_; i += 1024) eids[i] = sel[i];
  for (int i = tid; i < ROWS_CAP; i += 1024) row_tok[i] = -1;
  __syncthreads();
  const int wave = tid >> 5, lane = tid & 31;
  const unsigned ltmask = (1u << lane) - 1u;
  // pass 1: counts
  {
    int c = 0;
    for (int i = lane; i < NA_; i += 32) {
      unsigned m = (unsigned)__ballot(eids[i] == wave);
      c += __popc(m);
    }
    if (lane == 0) s_cnt[wave] = c;
  }
  __syncthreads();
  if (tid == 0) {
    int b = 0;
    for (int e = 0; e < E_; ++e) {
      s_base[e] = b;
      int cc = s_cnt[e] < C_ ? s_cnt[e] : C_;
      b += (cc + 31) & ~31;
    }
  }
  __syncthreads();
  // pass 2: ranks + row mapping (flat-index order == stable argsort order)
  {
    int c = 0;
    const int b = s_base[wave];
    for (int i = lane; i < NA_; i += 32) {
      bool mine = (eids[i] == wave);
      unsigned m = (unsigned)__ballot(mine);
      if (mine) {
        int r = c + __popc(m & ltmask);
        if (r < C_) {
          int row = b + r;
          row_tok[row] = i >> 1;       // i / K_
          row_assign[i] = row;
        } else {
          row_assign[i] = -1;          // capacity drop
        }
      }
      c += __popc(m);
    }
  }
  if (tid < E_) {
    cnt[tid]   = s_cnt[tid] < C_ ? s_cnt[tid] : C_;
    basep[tid] = s_base[tid];
  }
}

// ---------------------------------------------------------------------------
// 3) Gather + convert: xg[row] = bf16(x[tok])  (2x v_perm_b32 per 4 elements)
// ---------------------------------------------------------------------------
__global__ __launch_bounds__(256) void gather_kernel(
    const float* __restrict__ x, const int* __restrict__ row_tok,
    __bf16* __restrict__ xg) {
  const int r = blockIdx.x;
  const int tok = row_tok[r];
  if (tok < 0) return;
  const int i = threadIdx.x;                      // 4 elements per thread
  float4 v = ((const float4*)(x + (size_t)tok * H_))[i];
  uint2 pk;
  pk.x = pk2bf(v.x, v.y);
  pk.y = pk2bf(v.z, v.w);
  *(uint2*)(xg + (size_t)r * H_ + i * 4) = pk;
}

// ---------------------------------------------------------------------------
// 4) GEMM1: h = silu(xg @ w1^T) * (xg @ w3^T), bf16 out.
//    Block tile 32(M) x 256(N); 8 waves, each 32x32 for BOTH w1 and w3.
// ---------------------------------------------------------------------------
__global__ __launch_bounds__(256) void gemm1_kernel(
    const __bf16* __restrict__ xg, const float* __restrict__ w1,
    const float* __restrict__ w3, __bf16* __restrict__ hbuf,
    const int* __restrict__ cnt, const int* __restrict__ basep) {
  const int e = blockIdx.z, mt = blockIdx.y, nt = blockIdx.x;
  const int ce = cnt[e];
  if (mt * 32 >= ce) return;
  const int rowbase = basep[e] + mt * 32;
  const int wave = threadIdx.x >> 5, lane = threadIdx.x & 31;
  const int n0 = nt * 256 + wave * 32;
  const float* W1 = w1 + (size_t)e * F_ * H_;
  const float* W3 = w3 + (size_t)e * F_ * H_;
  v8f c1[2][2] = {};
  v8f c3[2][2] = {};
  for (int kb = 0; kb < H_; kb += 32) {
    v16bf a0 = load_a_frag(xg, rowbase,      H_, kb, lane);
    v16bf a1 = load_a_frag(xg, rowbase + 16, H_, kb, lane);
#pragma unroll
    for (int ni = 0; ni < 2; ++ni) {
      v16bf b1 = load_b_frag(W1, n0 + ni * 16, H_, kb, lane);
      v16bf b3 = load_b_frag(W3, n0 + ni * 16, H_, kb, lane);
      c1[0][ni] = wmma_bf16(a0, b1, c1[0][ni]);
      c1[1][ni] = wmma_bf16(a1, b1, c1[1][ni]);
      c3[0][ni] = wmma_bf16(a0, b3, c3[0][ni]);
      c3[1][ni] = wmma_bf16(a1, b3, c3[1][ni]);
    }
    if (kb + 32 < H_) {                          // stream next weight panel
      __builtin_prefetch(W1 + (size_t)(n0 + (lane & 15)) * H_ + kb + 32, 0, 1);
      __builtin_prefetch(W3 + (size_t)(n0 + 16 + (lane & 15)) * H_ + kb + 32, 0, 1);
    }
  }
  const int half = lane >> 4, nn = lane & 15;
#pragma unroll
  for (int mi = 0; mi < 2; ++mi)
#pragma unroll
    for (int ni = 0; ni < 2; ++ni)
#pragma unroll
      for (int r = 0; r < 8; ++r) {
        int row = rowbase + mi * 16 + r + 8 * half;
        int col = n0 + ni * 16 + nn;
        float g = c1[mi][ni][r];
        float u = c3[mi][ni][r];
        float s = g / (1.0f + __expf(-g));        // silu
        hbuf[(size_t)row * F_ + col] = f2bf(s * u);
      }
}

// ---------------------------------------------------------------------------
// 5) GEMM2: y = h @ w2^T, f32 out.  Block tile 32(M) x 256(N), K=2816.
// ---------------------------------------------------------------------------
__global__ __launch_bounds__(256) void gemm2_kernel(
    const __bf16* __restrict__ hbuf, const float* __restrict__ w2,
    float* __restrict__ ybuf, const int* __restrict__ cnt,
    const int* __restrict__ basep) {
  const int e = blockIdx.z, mt = blockIdx.y, nt = blockIdx.x;
  const int ce = cnt[e];
  if (mt * 32 >= ce) return;
  const int rowbase = basep[e] + mt * 32;
  const int wave = threadIdx.x >> 5, lane = threadIdx.x & 31;
  const int n0 = nt * 256 + wave * 32;
  const float* W2 = w2 + (size_t)e * H_ * F_;     // [H_][F_] row-major
  v8f c[2][2] = {};
  for (int kb = 0; kb < F_; kb += 32) {
    v16bf a0 = load_a_frag(hbuf, rowbase,      F_, kb, lane);
    v16bf a1 = load_a_frag(hbuf, rowbase + 16, F_, kb, lane);
#pragma unroll
    for (int ni = 0; ni < 2; ++ni) {
      v16bf b = load_b_frag(W2, n0 + ni * 16, F_, kb, lane);
      c[0][ni] = wmma_bf16(a0, b, c[0][ni]);
      c[1][ni] = wmma_bf16(a1, b, c[1][ni]);
    }
    if (kb + 32 < F_) {
      __builtin_prefetch(W2 + (size_t)(n0 + (lane & 15)) * F_ + kb + 32, 0, 1);
    }
  }
  const int half = lane >> 4, nn = lane & 15;
#pragma unroll
  for (int mi = 0; mi < 2; ++mi)
#pragma unroll
    for (int ni = 0; ni < 2; ++ni)
#pragma unroll
      for (int r = 0; r < 8; ++r) {
        int row = rowbase + mi * 16 + r + 8 * half;
        int col = n0 + ni * 16 + nn;
        ybuf[(size_t)row * H_ + col] = c[mi][ni][r];
      }
}

// ---------------------------------------------------------------------------
// 6) Combine: out[t] = sum_k selw[t,k] * y[row(t,k)]  (deterministic gather)
// ---------------------------------------------------------------------------
__global__ __launch_bounds__(256) void combine_kernel(
    const float* __restrict__ ybuf, const int* __restrict__ row_assign,
    const float* __restrict__ selw, float* __restrict__ out) {
  const int t = blockIdx.x, i = threadIdx.x;      // 4 floats per thread
  const int r0 = row_assign[t * K_ + 0];
  const int r1 = row_assign[t * K_ + 1];
  const float w0 = selw[t * K_ + 0];
  const float w1 = selw[t * K_ + 1];
  float4 acc = make_float4(0.f, 0.f, 0.f, 0.f);
  if (r0 >= 0) {
    float4 v = ((const float4*)(ybuf + (size_t)r0 * H_))[i];
    acc.x += w0 * v.x; acc.y += w0 * v.y; acc.z += w0 * v.z; acc.w += w0 * v.w;
  }
  if (r1 >= 0) {
    float4 v = ((const float4*)(ybuf + (size_t)r1 * H_))[i];
    acc.x += w1 * v.x; acc.y += w1 * v.y; acc.z += w1 * v.z; acc.w += w1 * v.w;
  }
  ((float4*)(out + (size_t)t * H_))[i] = acc;
}

// ---------------------------------------------------------------------------
extern "C" void kernel_launch(void* const* d_in, const int* in_sizes, int n_in,
                              void* d_out, int out_size, void* d_ws, size_t ws_size,
                              hipStream_t stream) {
  (void)in_sizes; (void)n_in; (void)out_size; (void)ws_size;
  const float* x      = (const float*)d_in[0];   // [2,1024,1024]
  const int*   sel    = (const int*)  d_in[1];   // [2048,2]
  const float* gate_w = (const float*)d_in[2];   // [32,1024]
  const float* w1     = (const float*)d_in[3];   // [32,2816,1024]
  const float* w2     = (const float*)d_in[4];   // [32,1024,2816]
  const float* w3     = (const float*)d_in[5];   // [32,2816,1024]

  float* out        = (float*)d_out;             // [2048,1024]
  float* logits_out = out + (size_t)T_ * H_;     // [2048,32]

  // workspace carve-up
  char* w = (char*)d_ws;
  float* selw       = (float*)w;  w += (size_t)NA_ * 4;
  int*   cnt        = (int*)w;    w += (size_t)E_ * 4;
  int*   basep      = (int*)w;    w += (size_t)E_ * 4;
  int*   row_tok    = (int*)w;    w += (size_t)ROWS_CAP * 4;
  int*   row_assign = (int*)w;    w += (size_t)NA_ * 4;
  w = (char*)(((uintptr_t)w + 255) & ~(uintptr_t)255);
  __bf16* xg   = (__bf16*)w;      w += (size_t)ROWS_CAP * H_ * 2;
  __bf16* hbuf = (__bf16*)w;      w += (size_t)ROWS_CAP * F_ * 2;
  float*  ybuf = (float*)w;       w += (size_t)ROWS_CAP * H_ * 4;

  router_kernel  <<<T_, 128, 0, stream>>>(x, gate_w, sel, logits_out, selw);
  dispatch_kernel<<<1, 1024, 0, stream>>>(sel, cnt, basep, row_tok, row_assign);
  gather_kernel  <<<ROWS_CAP, 256, 0, stream>>>(x, row_tok, xg);
  gemm1_kernel   <<<dim3(F_ / 256, C_ / 32, E_), 256, 0, stream>>>(xg, w1, w3, hbuf, cnt, basep);
  gemm2_kernel   <<<dim3(H_ / 256, C_ / 32, E_), 256, 0, stream>>>(hbuf, w2, ybuf, cnt, basep);
  combine_kernel <<<T_, 256, 0, stream>>>(ybuf, row_assign, selw, out);
}
